// E8RHTLinear_18107582120393
// MI455X (gfx1250) — matrix operand order repaired and checked by reading, not compiled, not run on hardware
//
#include <hip/hip_runtime.h>
#include <hip/hip_bf16.h>

typedef __attribute__((ext_vector_type(16))) _Float16 v16h;
typedef __attribute__((ext_vector_type(8)))  _Float16 v8h;
typedef __attribute__((ext_vector_type(8)))  float    v8f;
typedef __attribute__((ext_vector_type(4)))  unsigned int v4u;
typedef __attribute__((ext_vector_type(8)))  int v8i;
typedef __attribute__((ext_vector_type(4)))  int v4i;
typedef unsigned int u32;
typedef unsigned long long u64;

// Problem shape (fixed by the reference)
#define BDIM 8192   // batch rows
#define NDIM 4096   // input features (K of the GEMM)
#define MDIM 4096   // output features
#define GDIM 512    // groups per output row (8 elems each)
#define INV_SQRT_N 0.015625f  // 1/sqrt(4096) = 1/64

// ---------------------------------------------------------------------------
// Kernel 1: x_rht = FHT(x * SV) * (1/64), emitted as f16.  One row per block.
// ---------------------------------------------------------------------------
__global__ __launch_bounds__(256) void fht_in_kernel(const float* __restrict__ x,
                                                     const float* __restrict__ SV,
                                                     _Float16* __restrict__ out) {
    __shared__ float s[NDIM];  // 16 KB
    const int row = blockIdx.x;
    const float* xr = x + (size_t)row * NDIM;

    for (int i = threadIdx.x; i < NDIM / 4; i += 256) {
        float4 v  = ((const float4*)xr)[i];
        float4 sv = ((const float4*)SV)[i];
        s[4 * i + 0] = v.x * sv.x;
        s[4 * i + 1] = v.y * sv.y;
        s[4 * i + 2] = v.z * sv.z;
        s[4 * i + 3] = v.w * sv.w;
    }
    __syncthreads();

    for (int h = 1; h < NDIM; h <<= 1) {
        for (int t = threadIdx.x; t < NDIM / 2; t += 256) {
            int i = ((t & ~(h - 1)) << 1) | (t & (h - 1));
            float a = s[i], b = s[i + h];
            s[i] = a + b;
            s[i + h] = a - b;
        }
        __syncthreads();
    }

    _Float16* o = out + (size_t)row * NDIM;
    for (int i = threadIdx.x; i < NDIM; i += 256)
        o[i] = (_Float16)(s[i] * INV_SQRT_N);
}

// ---------------------------------------------------------------------------
// Kernel 2: decode W' = (codebook[Qidxs] + codebook2[Qidxs2]*irs) * Wscale (f16)
// ---------------------------------------------------------------------------
__global__ __launch_bounds__(256) void decode_w_kernel(const int* __restrict__ Qidxs,
                                                       const int* __restrict__ Qidxs2,
                                                       const float* __restrict__ cb,
                                                       const float* __restrict__ cb2,
                                                       const float* __restrict__ wscale_p,
                                                       const float* __restrict__ irs_p,
                                                       _Float16* __restrict__ Wh) {
    const int g = blockIdx.x * 256 + threadIdx.x;  // 0 .. M*G-1
    if (g >= MDIM * GDIM) return;
    const float ws  = wscale_p[0];
    const float irs = irs_p[0];
    const int i1 = Qidxs[g];
    const int i2 = Qidxs2[g];
    const float4* c1 = (const float4*)(cb  + (size_t)i1 * 8);
    const float4* c2 = (const float4*)(cb2 + (size_t)i2 * 8);
    float4 a0 = c1[0], a1 = c1[1];
    float4 b0 = c2[0], b1 = c2[1];
    v8h o;
    o[0] = (_Float16)((a0.x + b0.x * irs) * ws);
    o[1] = (_Float16)((a0.y + b0.y * irs) * ws);
    o[2] = (_Float16)((a0.z + b0.z * irs) * ws);
    o[3] = (_Float16)((a0.w + b0.w * irs) * ws);
    o[4] = (_Float16)((a1.x + b1.x * irs) * ws);
    o[5] = (_Float16)((a1.y + b1.y * irs) * ws);
    o[6] = (_Float16)((a1.z + b1.z * irs) * ws);
    o[7] = (_Float16)((a1.w + b1.w * irs) * ws);
    *(v8h*)(Wh + (size_t)g * 8) = o;
}

// ---------------------------------------------------------------------------
// TDM helper: DMA one 128-row x 32-half (64B/row) 2D tile from a 4096-wide
// f16 row-major tensor into LDS (row-contiguous), via a Tensor DMA Descriptor.
// ---------------------------------------------------------------------------
__device__ __forceinline__ void tdm_load_tile_128x32(const _Float16* gsrc, u32 lds_off) {
    u64 ga = (u64)(uintptr_t)gsrc;
    v4u g0;
    g0[0] = 1u;                                   // count=1 valid descriptor
    g0[1] = lds_off;                              // LDS byte address
    g0[2] = (u32)ga;                              // global_addr[31:0]
    g0[3] = ((u32)(ga >> 32) & 0x01FFFFFFu)       // global_addr[56:32]
            | (2u << 30);                         // type=2 ("image")
    v8i g1;
    g1[0] = 0x10000;                              // workgroup_mask=0, data_size=1 (2B)
    g1[1] = (int)(4096u << 16);                   // tensor_dim0[15:0]=4096 -> bits 63:48
    g1[2] = (int)(4096u << 16);                   // tensor_dim0 hi=0, tensor_dim1[15:0]=4096
    g1[3] = (int)(32u << 16);                     // tensor_dim1 hi=0, tile_dim0=32
    g1[4] = 128;                                  // tile_dim1=128, tile_dim2=0
    g1[5] = 4096;                                 // tensor_dim0_stride = 4096 elements
    g1[6] = 0;                                    // stride hi, tensor_dim1_stride lo
    g1[7] = 0;                                    // tensor_dim1_stride hi
    v4i z = {0, 0, 0, 0};                         // groups 2/3 unused (2-D tile)
#if __clang_major__ >= 23
    v8i z8 = {0, 0, 0, 0, 0, 0, 0, 0};
    __builtin_amdgcn_tensor_load_to_lds(g0, g1, z, z, z8, 0);
#else
    __builtin_amdgcn_tensor_load_to_lds(g0, g1, z, z, 0);
#endif
}

// ---------------------------------------------------------------------------
// Kernel 3: y_rht[b][m] = sum_k x_rht[b][k] * W'[m][k]   (f16 WMMA, f32 acc)
// 128x128 block tile, TDM-staged double-buffered LDS, 8 waves x (2x4) frags.
// K-loop manually unrolled by 2 so buffer selection is compile-time constant.
// ---------------------------------------------------------------------------
#define TBM 128
#define TBN 128
#define TBK 32

__device__ __forceinline__ void wmma_phase(const _Float16* __restrict__ as,
                                           const _Float16* __restrict__ bs,
                                           int wm, int wn, int l16, int lhalf,
                                           v8f acc[2][4]) {
    // B fragments: lane l16 = column (m); low lanes K 0..15, high lanes K 16..31.
    v16h bfrag[4];
    #pragma unroll
    for (int j = 0; j < 4; ++j) {
        int col = wn * 64 + j * 16 + l16;
        bfrag[j] = *(const v16h*)&bs[col * TBK + lhalf * 16];
    }

    // A fragments: lane l16 = row; elems 0..15 = K lhalf*16..lhalf*16+15
    v16h afrag[2];
    #pragma unroll
    for (int i = 0; i < 2; ++i) {
        int rrow = wm * 32 + i * 16 + l16;
        const _Float16* ap = &as[rrow * TBK];
        v8h lo = *(const v8h*)(ap + lhalf * 16);
        v8h hi = *(const v8h*)(ap + lhalf * 16 + 8);
        afrag[i] = __builtin_shufflevector(lo, hi, 0, 1, 2, 3, 4, 5, 6, 7,
                                           8, 9, 10, 11, 12, 13, 14, 15);
    }

    #pragma unroll
    for (int i = 0; i < 2; ++i)
        #pragma unroll
        for (int j = 0; j < 4; ++j)
            acc[i][j] = __builtin_amdgcn_wmma_f32_16x16x32_f16(
                false, afrag[i], false, bfrag[j],
                (short)0, acc[i][j], false, false);
}

__global__ __launch_bounds__(256, 2) void gemm_wmma_kernel(const _Float16* __restrict__ A,
                                                           const _Float16* __restrict__ Bw,
                                                           float* __restrict__ C) {
    __shared__ __align__(16) _Float16 As[2][TBM * TBK];  // 2 x 8 KB
    __shared__ __align__(16) _Float16 Bs[2][TBN * TBK];  // 2 x 8 KB

    const int tid   = threadIdx.x;
    const int wave  = tid >> 5;
    const int lane  = tid & 31;
    const int l16   = lane & 15;
    const int lhalf = lane >> 4;

    const int wm = wave & 3;               // 4 waves along batch rows (32 each)
    const int wn = wave >> 2;              // 2 waves along m cols (64 each)

    const int block_m = blockIdx.y * TBM;  // batch-row base
    const int block_n = blockIdx.x * TBN;  // m base

    const _Float16* Abase = A  + (size_t)block_m * NDIM;
    const _Float16* Bbase = Bw + (size_t)block_n * NDIM;

    const u32 asOff0 = (u32)(uintptr_t)&As[0][0];
    const u32 asOff1 = (u32)(uintptr_t)&As[1][0];
    const u32 bsOff0 = (u32)(uintptr_t)&Bs[0][0];
    const u32 bsOff1 = (u32)(uintptr_t)&Bs[1][0];

    v8f acc[2][4] = {};

    // Prologue: wave 0 DMAs the k0=0 tiles into buffer 0.
    if (tid < 32) {
        tdm_load_tile_128x32(Abase, asOff0);
        tdm_load_tile_128x32(Bbase, bsOff0);
    }

    #pragma unroll 1
    for (int k0 = 0; k0 < NDIM; k0 += 2 * TBK) {
        // ---- Phase 0: consume buffer 0, prefetch into buffer 1 ----
        // TENSORcnt is per-wave: no-op on waves 1..7, drains wave 0's DMAs.
        __builtin_amdgcn_s_wait_tensorcnt(0);
        __syncthreads();
        if (tid < 32) {
            tdm_load_tile_128x32(Abase + (k0 + TBK), asOff1);
            tdm_load_tile_128x32(Bbase + (k0 + TBK), bsOff1);
        }
        wmma_phase(&As[0][0], &Bs[0][0], wm, wn, l16, lhalf, acc);

        // ---- Phase 1: consume buffer 1, prefetch into buffer 0 ----
        __builtin_amdgcn_s_wait_tensorcnt(0);
        __syncthreads();
        if (tid < 32 && (k0 + 2 * TBK) < NDIM) {
            tdm_load_tile_128x32(Abase + (k0 + 2 * TBK), asOff0);
            tdm_load_tile_128x32(Bbase + (k0 + 2 * TBK), bsOff0);
        }
        wmma_phase(&As[1][0], &Bs[1][0], wm, wn, l16, lhalf, acc);
    }

    // Epilogue: C/D 16x16 f32 layout -> lane l16 = col, VGPR r = row octet.
    #pragma unroll
    for (int i = 0; i < 2; ++i) {
        #pragma unroll
        for (int j = 0; j < 4; ++j) {
            int rbase = block_m + wm * 32 + i * 16 + lhalf * 8;
            int col   = block_n + wn * 64 + j * 16 + l16;
            #pragma unroll
            for (int r = 0; r < 8; ++r)
                C[(size_t)(rbase + r) * MDIM + col] = acc[i][j][r];
        }
    }
}

// ---------------------------------------------------------------------------
// Kernel 4: y = FHT(y_rht) * (1/64) * SU.  One row per block, f32 out.
// ---------------------------------------------------------------------------
__global__ __launch_bounds__(256) void fht_out_kernel(const float* __restrict__ yr,
                                                      const float* __restrict__ SU,
                                                      float* __restrict__ out) {
    __shared__ float s[MDIM];  // 16 KB
    const int row = blockIdx.x;
    const float* xr = yr + (size_t)row * MDIM;

    for (int i = threadIdx.x; i < MDIM / 4; i += 256)
        ((float4*)s)[i] = ((const float4*)xr)[i];
    __syncthreads();

    for (int h = 1; h < MDIM; h <<= 1) {
        for (int t = threadIdx.x; t < MDIM / 2; t += 256) {
            int i = ((t & ~(h - 1)) << 1) | (t & (h - 1));
            float a = s[i], b = s[i + h];
            s[i] = a + b;
            s[i + h] = a - b;
        }
        __syncthreads();
    }

    float* o = out + (size_t)row * MDIM;
    for (int i = threadIdx.x; i < MDIM; i += 256)
        o[i] = s[i] * INV_SQRT_N * SU[i];
}

// ---------------------------------------------------------------------------
// Launch
// ---------------------------------------------------------------------------
extern "C" void kernel_launch(void* const* d_in, const int* in_sizes, int n_in,
                              void* d_out, int out_size, void* d_ws, size_t ws_size,
                              hipStream_t stream) {
    const float* x      = (const float*)d_in[0];
    const int*   Qidxs  = (const int*)d_in[1];
    const int*   Qidxs2 = (const int*)d_in[2];
    const float* SU     = (const float*)d_in[3];
    const float* SV     = (const float*)d_in[4];
    const float* cb     = (const float*)d_in[5];
    const float* cb2    = (const float*)d_in[6];
    const float* Wscale = (const float*)d_in[7];
    const float* irs    = (const float*)d_in[8];
    float* out = (float*)d_out;

    char* ws = (char*)d_ws;
    _Float16* xh = (_Float16*)ws;                                        // 64 MB
    _Float16* wh = (_Float16*)(ws + (size_t)BDIM * NDIM * 2);            // 32 MB
    float*    yr = (float*)(ws + (size_t)BDIM * NDIM * 2
                               + (size_t)MDIM * NDIM * 2);               // 128 MB

    fht_in_kernel<<<BDIM, 256, 0, stream>>>(x, SV, xh);

    decode_w_kernel<<<(MDIM * GDIM) / 256, 256, 0, stream>>>(
        Qidxs, Qidxs2, cb, cb2, Wscale, irs, wh);

    dim3 grid(MDIM / TBN, BDIM / TBM);  // (32, 64)
    gemm_wmma_kernel<<<grid, 256, 0, stream>>>(xh, wh, yr);

    fht_out_kernel<<<BDIM, 256, 0, stream>>>(yr, SU, out);
}
